// NemotronFlashAttention_7292854468787
// MI455X (gfx1250) — matrix operand order repaired and checked by reading, not compiled, not run on hardware
//
#include <hip/hip_runtime.h>
#include <hip/hip_bf16.h>

typedef __attribute__((ext_vector_type(16))) _Float16 v16h;
typedef __attribute__((ext_vector_type(8)))  _Float16 v8h;
typedef __attribute__((ext_vector_type(4)))  _Float16 v4h;
typedef __attribute__((ext_vector_type(8)))  float    v8f;
typedef __attribute__((ext_vector_type(4)))  unsigned int v4u;
typedef __attribute__((ext_vector_type(8)))  int      v8i;
typedef __attribute__((ext_vector_type(4)))  int      v4i;

#define DEV __device__ __forceinline__

static constexpr int B_  = 2;
static constexpr int S_  = 2048;
static constexpr int D_  = 2048;
static constexpr int H_  = 16;
static constexpr int KV_ = 4;
static constexpr int DQK = 96;
static constexpr int DV  = 128;

// ---------------- helpers ----------------

DEV v8h cvt8(float4 a, float4 b, float s) {
    v8h h;
    h[0] = (_Float16)(a.x * s); h[1] = (_Float16)(a.y * s);
    h[2] = (_Float16)(a.z * s); h[3] = (_Float16)(a.w * s);
    h[4] = (_Float16)(b.x * s); h[5] = (_Float16)(b.y * s);
    h[6] = (_Float16)(b.z * s); h[7] = (_Float16)(b.w * s);
    return h;
}

DEV v16h cat16(v8h lo, v8h hi) {
    return __builtin_shufflevector(lo, hi, 0,1,2,3,4,5,6,7,8,9,10,11,12,13,14,15);
}

// A-fragment (16x32 f16): lane half -> kbase 0/8; elems 0..7 = K kb..kb+7,
// elems 8..15 = K kb+16..kb+23.  Both runs are 8 contiguous halves.
DEV v16h load_a_lds(const _Float16* p) {   // p = &row[kbase]
    v8h lo = *(const v8h*)(p);
    v8h hi = *(const v8h*)(p + 16);
    return cat16(lo, hi);
}

DEV float rowmax16(float x) {
#pragma unroll
    for (int m = 1; m < 16; m <<= 1) x = fmaxf(x, __shfl_xor(x, m, 32));
    return x;
}
DEV float rowsum16(float x) {
#pragma unroll
    for (int m = 1; m < 16; m <<= 1) x += __shfl_xor(x, m, 32);
    return x;
}

// Issue a TDM 2D tile load: rows x 32 f16 elements into LDS.
// D# packing per CDNA5 ISA.
// This toolchain exposes the 6-arg builtin:
//   (uint32x4 g0, int32x8 g1, int32x4 g2, int32x4 g3, int32x8 pad, i32 cpol)
DEV void tdm_tile_load(unsigned lds_addr, const _Float16* gptr,
                       int rows, int tensor_rows, int K) {
#if __has_builtin(__builtin_amdgcn_tensor_load_to_lds)
    unsigned long long ga = (unsigned long long)(uintptr_t)gptr;
    v4u g0;
    g0[0] = 1u;                                   // count=1, user mode
    g0[1] = lds_addr;                             // D#.lds_addr (LDS byte offset)
    g0[2] = (unsigned)(ga & 0xFFFFFFFFull);       // global_addr[31:0]
    g0[3] = (unsigned)(ga >> 32) | (2u << 30);    // global_addr[56:32] | type=2
    v8i g1;
    g1[0] = 0x00010000;                           // workgroup_mask=0, data_size=1 (2B)
    g1[1] = (K & 0xFFFF) << 16;                   // tensor_dim0[15:0]
    g1[2] = ((K >> 16) & 0xFFFF) | ((tensor_rows & 0xFFFF) << 16);
    g1[3] = ((tensor_rows >> 16) & 0xFFFF) | (32 << 16);   // tile_dim0 = 32
    g1[4] = rows & 0xFFFF;                        // tile_dim1 = rows, tile_dim2 = 0
    g1[5] = K;                                    // tensor_dim0_stride[31:0]
    g1[6] = 0;
    g1[7] = 0;
    v4i z4 = {0, 0, 0, 0};
    v8i z8 = {0, 0, 0, 0, 0, 0, 0, 0};
    __builtin_amdgcn_tensor_load_to_lds(g0, g1, z4, z4, z8, 0);
#endif
}

// ---------------- f32 -> f16 elementwise convert ----------------------------

__global__ __launch_bounds__(256)
void cvt_f32_f16(const float* __restrict__ in, _Float16* __restrict__ out) {
    const size_t i = ((size_t)blockIdx.x * 256 + threadIdx.x) * 8;
    float4 a = *(const float4*)(in + i);
    float4 b = *(const float4*)(in + i + 4);
    *(v8h*)(out + i) = cvt8(a, b, 1.f);
}

// ---------------- GEMM: C[M,N] = X[M,K] * W[N,K]^T  (f16 in, f32 out) -------

__global__ __launch_bounds__(256)
void gemm_xwT(const _Float16* __restrict__ X, const _Float16* __restrict__ W,
              float* __restrict__ C, int M, int N, int K) {
    __shared__ __attribute__((aligned(64))) _Float16 Xs[128 * 32];
    __shared__ __attribute__((aligned(64))) _Float16 Ws[64 * 32];

    const int tid  = threadIdx.x;
    const int lane = tid & 31;
    const int wave = tid >> 5;
    const int wm   = wave >> 1;          // 0..3
    const int wn   = wave & 1;           // 0..1
    const int bm   = blockIdx.y * 128;
    const int bn   = blockIdx.x * 64;

    v8f acc[2][2] = {};
    const int kbA = (lane < 16) ? 0 : 8;
    const int kbB = (lane < 16) ? 0 : 16;
    const int m15 = lane & 15;

#if __has_builtin(__builtin_amdgcn_tensor_load_to_lds)
    const unsigned ldsX = (unsigned)(uintptr_t)&Xs[0];
    const unsigned ldsW = (unsigned)(uintptr_t)&Ws[0];
#endif

    for (int k0 = 0; k0 < K; k0 += 32) {
#if __has_builtin(__builtin_amdgcn_tensor_load_to_lds)
        if (wave == 0) {                 // TDM ignores EXEC; one issue per tile
            tdm_tile_load(ldsX, X + (size_t)bm * K + k0, 128, M, K);
            tdm_tile_load(ldsW, W + (size_t)bn * K + k0, 64,  N, K);
            __builtin_amdgcn_s_wait_tensorcnt(0);
        }
#else
        {   // manual staging fallback: straight f16 b128 copies
            const int xr = tid >> 1, xc = (tid & 1) * 16;
            *(v16h*)&Xs[xr * 32 + xc] =
                *(const v16h*)(X + (size_t)(bm + xr) * K + k0 + xc);
            const int wr = tid >> 2, wc = (tid & 3) * 8;
            *(v8h*)&Ws[wr * 32 + wc] =
                *(const v8h*)(W + (size_t)(bn + wr) * K + k0 + wc);
        }
#endif
        __syncthreads();

        v16h a[2], b[2];
#pragma unroll
        for (int mi = 0; mi < 2; ++mi) {
            int row = wm * 32 + mi * 16 + m15;
            a[mi] = load_a_lds(&Xs[row * 32 + kbA]);
        }
#pragma unroll
        for (int ni = 0; ni < 2; ++ni) {
            int col = wn * 32 + ni * 16 + m15;
            b[ni] = load_a_lds(&Ws[col * 32 + kbA]);
        }
#pragma unroll
        for (int mi = 0; mi < 2; ++mi)
#pragma unroll
            for (int ni = 0; ni < 2; ++ni)
                acc[mi][ni] = __builtin_amdgcn_wmma_f32_16x16x32_f16(
                    false, a[mi], false, b[ni], (short)0, acc[mi][ni], false, false);
        __syncthreads();
    }

    const int rbase = (lane >> 4) << 3;   // 0 or 8
    const int coll  = lane & 15;
#pragma unroll
    for (int mi = 0; mi < 2; ++mi)
#pragma unroll
        for (int ni = 0; ni < 2; ++ni)
#pragma unroll
            for (int r = 0; r < 8; ++r)
                C[(size_t)(bm + wm * 32 + mi * 16 + rbase + r) * N +
                  bn + wn * 32 + ni * 16 + coll] = acc[mi][ni][r];
}

// ---------------- RMSNorm + RoPE (in place on q and k) ----------------------
// one wave per (b, s, head);  heads 0..H-1 -> q, H..H+KV-1 -> k

__global__ __launch_bounds__(32)
void rmsnorm_rope(float* __restrict__ q, float* __restrict__ k,
                  const float* __restrict__ qw, const float* __restrict__ kw,
                  const int* __restrict__ pos_ids) {
    const int idx = blockIdx.x;
    const int hh  = idx % (H_ + KV_);
    const int s   = (idx / (H_ + KV_)) % S_;
    const int b   = idx / ((H_ + KV_) * S_);

    float* x;
    const float* w;
    if (hh < H_) { x = q + (((size_t)(b * S_ + s)) * H_ + hh) * DQK;        w = qw; }
    else         { x = k + (((size_t)(b * S_ + s)) * KV_ + (hh - H_)) * DQK; w = kw; }

    const int t = threadIdx.x;                 // 0..31
    const int j0 = t, j1 = t + 32, j2 = t + 64;
    const int p0 = t + 48;
    const int p1 = (t < 16) ? t + 80 : t - 16;
    const int p2 = t + 16;

    float x0 = x[j0], x1 = x[j1], x2 = x[j2];
    float y0 = x[p0], y1 = x[p1], y2 = x[p2];
    float w0 = w[j0], w1 = w[j1], w2 = w[j2];
    float u0 = w[p0], u1 = w[p1], u2 = w[p2];

    float ss = x0 * x0 + x1 * x1 + x2 * x2;
#pragma unroll
    for (int m = 1; m < 32; m <<= 1) ss += __shfl_xor(ss, m, 32);
    const float inv = rsqrtf(ss * (1.0f / 96.0f) + 1e-6f);
    const float pos = (float)pos_ids[b * S_ + s];

    float r0, r1, r2;
    {   int i = j0 % 48;
        float f = pos * __powf(10000.0f, -(float)i / 48.0f);
        float c = __cosf(f), sn = __sinf(f);
        float nj = x0 * inv * w0, np = y0 * inv * u0;
        r0 = (j0 < 48) ? nj * c - np * sn : nj * c + np * sn; }
    {   int i = j1 % 48;
        float f = pos * __powf(10000.0f, -(float)i / 48.0f);
        float c = __cosf(f), sn = __sinf(f);
        float nj = x1 * inv * w1, np = y1 * inv * u1;
        r1 = (j1 < 48) ? nj * c - np * sn : nj * c + np * sn; }
    {   int i = j2 % 48;
        float f = pos * __powf(10000.0f, -(float)i / 48.0f);
        float c = __cosf(f), sn = __sinf(f);
        float nj = x2 * inv * w2, np = y2 * inv * u2;
        r2 = nj * c + np * sn; }                // j2 >= 64 > 48 always

    x[j0] = r0; x[j1] = r1; x[j2] = r2;
}

// ---------------- causal flash attention (GQA) ------------------------------
// block = 128 threads (4 waves); each block: one (b, h), 64 query rows.
// ctx is written as f16 so the O-projection GEMM consumes it directly.

__global__ __launch_bounds__(128)
void flash_attn(const float* __restrict__ q, const float* __restrict__ k,
                const float* __restrict__ v, _Float16* __restrict__ ctx) {
    __shared__ __attribute__((aligned(64))) _Float16 Ks[32 * DQK];     // [key][d]
    __shared__ __attribute__((aligned(64))) _Float16 VsT[DV * 32];     // [vd][key]
    __shared__ __attribute__((aligned(64))) _Float16 Ps[4][16 * 32];   // per wave

    const int idx = blockIdx.x;
    const int qb  = idx % (S_ / 64);
    const int h   = (idx / (S_ / 64)) % H_;
    const int b   = idx / ((S_ / 64) * H_);
    const int kvh = h >> 2;
    const int q0  = qb * 64;

    const int tid  = threadIdx.x;
    const int lane = tid & 31;
    const int w    = tid >> 5;
    const int m15  = lane & 15;
    const int kbA  = (lane < 16) ? 0 : 8;
    const int kbB  = (lane < 16) ? 0 : 16;
    const int rbase = (lane >> 4) << 3;

    // Q A-fragments (scale folded in), 3 chunks of K=32 over d=96
    const float scale = 0.1020620726f;          // 1/sqrt(96)
    v16h qa[3];
    {
        const int qrow = q0 + w * 16 + m15;
        const float* qp = q + ((size_t)(b * S_ + qrow) * H_ + h) * DQK;
#pragma unroll
        for (int c = 0; c < 3; ++c) {
            const float* p = qp + 32 * c + kbA;
            float4 f0 = *(const float4*)(p);
            float4 f1 = *(const float4*)(p + 4);
            float4 f2 = *(const float4*)(p + 16);
            float4 f3 = *(const float4*)(p + 20);
            qa[c] = cat16(cvt8(f0, f1, scale), cvt8(f2, f3, scale));
        }
    }

    v8f accO[8] = {};
    float mrow[8], lrow[8];
#pragma unroll
    for (int r = 0; r < 8; ++r) { mrow[r] = -__builtin_inff(); lrow[r] = 0.0f; }

    const int wave_qmax = q0 + w * 16 + 15;
    const int ntiles = q0 / 32 + 2;             // keys < q0 + 64

    const int ldr = tid >> 2;                   // 0..31: key row for staging
    const int kc0 = (tid & 3) * 24;             // K-tile cols
    const int vc0 = (tid & 3) * 32;             // V-tile cols

    for (int kt = 0; kt < ntiles; ++kt) {
        const int kpos = kt * 32;
        { // stage K tile [32 x 96] as f16
            const float* kp = k + ((size_t)(b * S_ + kpos + ldr) * KV_ + kvh) * DQK + kc0;
#pragma unroll
            for (int j = 0; j < 6; ++j) {
                float4 f = *(const float4*)(kp + j * 4);
                v4h hv; hv[0]=(_Float16)f.x; hv[1]=(_Float16)f.y;
                        hv[2]=(_Float16)f.z; hv[3]=(_Float16)f.w;
                *(v4h*)&Ks[ldr * DQK + kc0 + j * 4] = hv;
            }
        }
        { // stage V tile transposed: VsT[vd][key]
            const float* vp = v + ((size_t)(b * S_ + kpos + ldr) * KV_ + kvh) * DV + vc0;
#pragma unroll
            for (int j = 0; j < 8; ++j) {
                float4 f = *(const float4*)(vp + j * 4);
                VsT[(vc0 + j * 4 + 0) * 32 + ldr] = (_Float16)f.x;
                VsT[(vc0 + j * 4 + 1) * 32 + ldr] = (_Float16)f.y;
                VsT[(vc0 + j * 4 + 2) * 32 + ldr] = (_Float16)f.z;
                VsT[(vc0 + j * 4 + 3) * 32 + ldr] = (_Float16)f.w;
            }
        }
        __syncthreads();

        if (kpos <= wave_qmax) {                // wave-uniform: EXEC stays full
            // scores: Q(16x96) . K^T -> 16x32 in two accumulators
            v8f s0a = {}, s1a = {};
#pragma unroll
            for (int c = 0; c < 3; ++c) {
                v16h b0 = load_a_lds(&Ks[m15 * DQK + 32 * c + kbA]);
                v16h b1 = load_a_lds(&Ks[(m15 + 16) * DQK + 32 * c + kbA]);
                s0a = __builtin_amdgcn_wmma_f32_16x16x32_f16(false, qa[c], false, b0,
                                                             (short)0, s0a, false, false);
                s1a = __builtin_amdgcn_wmma_f32_16x16x32_f16(false, qa[c], false, b1,
                                                             (short)0, s1a, false, false);
            }
            // online softmax over the 32 new columns
#pragma unroll
            for (int r = 0; r < 8; ++r) {
                const int rowq = q0 + w * 16 + rbase + r;
                const int c0p = kpos + m15, c1p = c0p + 16;
                float s0 = (c0p <= rowq) ? s0a[r] : -__builtin_inff();
                float s1 = (c1p <= rowq) ? s1a[r] : -__builtin_inff();
                float mnew = fmaxf(mrow[r], rowmax16(fmaxf(s0, s1)));
                float p0 = __expf(s0 - mnew);
                float p1 = __expf(s1 - mnew);
                float rs = rowsum16(p0 + p1);
                float alpha = __expf(mrow[r] - mnew);
                lrow[r] = lrow[r] * alpha + rs;
                mrow[r] = mnew;
#pragma unroll
                for (int n = 0; n < 8; ++n) accO[n][r] *= alpha;
                Ps[w][(rbase + r) * 32 + m15]      = (_Float16)p0;
                Ps[w][(rbase + r) * 32 + m15 + 16] = (_Float16)p1;
            }
            asm volatile("s_wait_dscnt 0" ::: "memory");
            // P(16x32) . V(32x128)
            v16h pa = load_a_lds(&Ps[w][m15 * 32 + kbA]);
#pragma unroll
            for (int n = 0; n < 8; ++n) {
                v16h vb = load_a_lds(&VsT[(n * 16 + m15) * 32 + kbA]);
                accO[n] = __builtin_amdgcn_wmma_f32_16x16x32_f16(false, pa, false, vb,
                                                                 (short)0, accO[n], false, false);
            }
        }
        __syncthreads();
    }

    // epilogue: normalize and write ctx[b, row, h*128 + col] as f16
    float rinv[8];
#pragma unroll
    for (int r = 0; r < 8; ++r) rinv[r] = 1.0f / lrow[r];
#pragma unroll
    for (int n = 0; n < 8; ++n)
#pragma unroll
        for (int r = 0; r < 8; ++r) {
            const int rowq = q0 + w * 16 + rbase + r;
            ctx[(size_t)(b * S_ + rowq) * (H_ * DV) + h * DV + n * 16 + m15] =
                (_Float16)(accO[n][r] * rinv[r]);
        }
}

// ---------------- launch ----------------------------------------------------

extern "C" void kernel_launch(void* const* d_in, const int* in_sizes, int n_in,
                              void* d_out, int out_size, void* d_ws, size_t ws_size,
                              hipStream_t stream) {
    const float* hs  = (const float*)d_in[0];
    const int*   pos = (const int*)  d_in[1];
    const float* Wq  = (const float*)d_in[2];
    const float* Wk  = (const float*)d_in[3];
    const float* Wv  = (const float*)d_in[4];
    const float* Wo  = (const float*)d_in[5];
    const float* qnw = (const float*)d_in[6];
    const float* knw = (const float*)d_in[7];
    float* out = (float*)d_out;

    const int M = B_ * S_;                                   // 4096
    char* ws = (char*)d_ws;
    size_t off = 0;
    float* qbuf = (float*)(ws + off); off += (size_t)M * H_  * DQK * 4;
    float* kbuf = (float*)(ws + off); off += (size_t)M * KV_ * DQK * 4;
    float* vbuf = (float*)(ws + off); off += (size_t)M * KV_ * DV  * 4;
    _Float16* ctx16 = (_Float16*)(ws + off); off += (size_t)M * D_ * 2;
    _Float16* hs16  = (_Float16*)(ws + off); off += (size_t)M * D_ * 2;
    _Float16* wq16  = (_Float16*)(ws + off); off += (size_t)(H_  * DQK) * D_ * 2;
    _Float16* wk16  = (_Float16*)(ws + off); off += (size_t)(KV_ * DQK) * D_ * 2;
    _Float16* wv16  = (_Float16*)(ws + off); off += (size_t)(KV_ * DV)  * D_ * 2;
    _Float16* wo16  = (_Float16*)(ws + off); off += (size_t)D_ * D_ * 2;

    // one-time f32 -> f16 conversion of GEMM operands (2048 elems per block)
    cvt_f32_f16<<<dim3((M * D_) / 2048),            256, 0, stream>>>(hs, hs16);
    cvt_f32_f16<<<dim3((H_ * DQK * D_) / 2048),     256, 0, stream>>>(Wq, wq16);
    cvt_f32_f16<<<dim3((KV_ * DQK * D_) / 2048),    256, 0, stream>>>(Wk, wk16);
    cvt_f32_f16<<<dim3((KV_ * DV * D_) / 2048),     256, 0, stream>>>(Wv, wv16);
    cvt_f32_f16<<<dim3((D_ * D_) / 2048),           256, 0, stream>>>(Wo, wo16);

    // QKV projections (f16 WMMA, TDM-staged tiles)
    gemm_xwT<<<dim3((H_ * DQK) / 64, M / 128), 256, 0, stream>>>(hs16, wq16, qbuf, M, H_ * DQK, D_);
    gemm_xwT<<<dim3((KV_ * DQK) / 64, M / 128), 256, 0, stream>>>(hs16, wk16, kbuf, M, KV_ * DQK, D_);
    gemm_xwT<<<dim3((KV_ * DV)  / 64, M / 128), 256, 0, stream>>>(hs16, wv16, vbuf, M, KV_ * DV, D_);

    // per-head RMSNorm + RoPE
    rmsnorm_rope<<<dim3(B_ * S_ * (H_ + KV_)), 32, 0, stream>>>(qbuf, kbuf, qnw, knw, pos);

    // causal flash attention -> ctx (f16)
    flash_attn<<<dim3(B_ * H_ * (S_ / 64)), 128, 0, stream>>>(qbuf, kbuf, vbuf, ctx16);

    // output projection
    gemm_xwT<<<dim3(D_ / 64, M / 128), 256, 0, stream>>>(ctx16, wo16, out, M, D_, D_);
}
